// PointerNetwork_68925635166913
// MI455X (gfx1250) — compile-verified, matrix operand order checked by reading, not run on hardware
//
#include <hip/hip_runtime.h>
#include <hip/hip_bf16.h>

// ---------------------------------------------------------------------------
// Pointer network on gfx1250: bf16 WMMA (v_wmma_f32_16x16x32_bf16) for all
// GEMMs, f32 accumulation, fused LSTM cell update per step, bf16-resident e1.
// Round 2: K-loop outermost with 4-tile A reuse (no A-frag scratch spills),
// 16-wave step kernels with K split in quarters + LDS reduction.
// ---------------------------------------------------------------------------

typedef __attribute__((ext_vector_type(16))) __bf16 v16bf;
typedef __attribute__((ext_vector_type(8)))  float  v8f;

union FragBF { v16bf v; uint4 q[2]; };

#define NEGV (-1000000000.0f)

__device__ __forceinline__ unsigned short f2bf(float f) {
    unsigned int u = __float_as_uint(f);
    unsigned int r = u + 0x7FFFu + ((u >> 16) & 1u);
    return (unsigned short)(r >> 16);
}
__device__ __forceinline__ float bf2f(unsigned short s) {
    return __uint_as_float(((unsigned int)s) << 16);
}
__device__ __forceinline__ float sigf(float x) {
    return 1.0f / (1.0f + __expf(-x));
}
__device__ __forceinline__ v8f wmma_bf(const FragBF& a, const FragBF& b, const v8f& c) {
    return __builtin_amdgcn_wmma_f32_16x16x32_bf16(false, a.v, false, b.v,
                                                   (short)0, c, false, false);
}

// ------------------------- init / conversion kernels -----------------------

__global__ void cvt_bf(unsigned short* __restrict__ dst, const float* __restrict__ src, size_t n) {
    for (size_t i = (size_t)blockIdx.x * blockDim.x + threadIdx.x; i < n;
         i += (size_t)gridDim.x * blockDim.x)
        dst[i] = f2bf(src[i]);
}

// dst[r, 0:ka] = bf16(a[r,:]); dst[r, ka:ka+kb] = bf16(b[r,:])
__global__ void cat_cvt(unsigned short* __restrict__ dst, const float* __restrict__ a,
                        const float* __restrict__ b, int rows, int ka, int kb) {
    int K = ka + kb;
    size_t total = (size_t)rows * K;
    for (size_t i = (size_t)blockIdx.x * blockDim.x + threadIdx.x; i < total;
         i += (size_t)gridDim.x * blockDim.x) {
        int r = (int)(i / K), k = (int)(i % K);
        float v = (k < ka) ? a[(size_t)r * ka + k] : b[(size_t)r * kb + (k - ka)];
        dst[i] = f2bf(v);
    }
}

__global__ void vadd_f32(float* __restrict__ d, const float* __restrict__ a,
                         const float* __restrict__ b, int n) {
    int i = blockIdx.x * blockDim.x + threadIdx.x;
    if (i < n) d[i] = a[i] + b[i];
}

__global__ void zero_u32(unsigned int* __restrict__ p, size_t n) {
    for (size_t i = (size_t)blockIdx.x * blockDim.x + threadIdx.x; i < n;
         i += (size_t)gridDim.x * blockDim.x)
        p[i] = 0u;
}

// zero the padded slot enc_ext[:, 512, :]  (enc_ext is [64, 513, 512] bf16)
__global__ void zero_pad(unsigned short* __restrict__ enc_ext) {
    int i = blockIdx.x * blockDim.x + threadIdx.x; // 0..64*512-1
    if (i < 64 * 512) {
        int b = i >> 9, j = i & 511;
        enc_ext[((size_t)b * 513 + 512) * 512 + j] = 0;
    }
}

// ----------------------- fused LSTM step (enc & dec) -----------------------
// grid 32 WGs x 512 thr (16 waves). WG owns hidden slice [16*wg, 16*wg+16).
// wave = (kquarter, rowTile): computes 4 gate tiles over K/4, LDS-reduced.
// A row b of [x | h]; X source differs between encoder and decoder.

#define STEP_BODY(KTOT, XROW_EXPR, XLIM)                                        \
    __shared__ float gbuf[4][4][64][16]; /* [kq][gate][b][jj] = 64 KB */        \
    int wg = blockIdx.x;                                                        \
    int tid = threadIdx.x;                                                      \
    int wave = tid >> 5, lane = tid & 31;                                       \
    int r = wave & 3, kq = wave >> 2;                                           \
    int ln = lane & 15, hi = lane >> 4;                                         \
    int m0 = r * 16;                                                            \
    int b = m0 + ln;                                                            \
    const int KQ = (KTOT) / 4;                                                  \
    int k_begin = kq * KQ;                                                      \
    const unsigned short* xrow = (XROW_EXPR);                                   \
    const unsigned short* hrow = hprev + (size_t)b * hstride;                   \
    const unsigned short* wb = W + (size_t)(wg * 16 + ln) * (KTOT);             \
    const size_t gstride = (size_t)512 * (KTOT);                                \
    v8f c0 = {}, c1 = {}, c2 = {}, c3 = {};                                     \
    _Pragma("unroll 1")                                                         \
    for (int kk = k_begin; kk < k_begin + KQ; kk += 32) {                       \
        int kb = kk + hi * 8;                                                   \
        FragBF a, b0, b1, b2, b3;                                               \
        a.q[0] = *(const uint4*)((kb < (XLIM)) ? (xrow + kb)                    \
                                               : (hrow + (kb - (XLIM))));       \
        a.q[1] = *(const uint4*)((kb + 16 < (XLIM)) ? (xrow + kb + 16)          \
                                               : (hrow + (kb + 16 - (XLIM)))); \
        b0.q[0] = *(const uint4*)(wb + kb);                                     \
        b0.q[1] = *(const uint4*)(wb + kb + 16);                                \
        b1.q[0] = *(const uint4*)(wb + gstride + kb);                           \
        b1.q[1] = *(const uint4*)(wb + gstride + kb + 16);                      \
        b2.q[0] = *(const uint4*)(wb + 2 * gstride + kb);                       \
        b2.q[1] = *(const uint4*)(wb + 2 * gstride + kb + 16);                  \
        b3.q[0] = *(const uint4*)(wb + 3 * gstride + kb);                       \
        b3.q[1] = *(const uint4*)(wb + 3 * gstride + kb + 16);                  \
        c0 = wmma_bf(a, b0, c0);                                                \
        c1 = wmma_bf(a, b1, c1);                                                \
        c2 = wmma_bf(a, b2, c2);                                                \
        c3 = wmma_bf(a, b3, c3);                                                \
    }                                                                           \
    _Pragma("unroll")                                                           \
    for (int p = 0; p < 8; ++p) {                                               \
        int M = m0 + (hi ? p + 8 : p);                                          \
        gbuf[kq][0][M][ln] = c0[p];                                             \
        gbuf[kq][1][M][ln] = c1[p];                                             \
        gbuf[kq][2][M][ln] = c2[p];                                             \
        gbuf[kq][3][M][ln] = c3[p];                                             \
    }                                                                           \
    __syncthreads();

#define CELL_UPDATE(H_STORE)                                                    \
    _Pragma("unroll")                                                           \
    for (int q = 0; q < 2; ++q) {                                               \
        int e = tid + 512 * q;                                                  \
        int bb = e >> 4, jj = e & 15;                                           \
        int jg = wg * 16 + jj;                                                  \
        float gi = 0.f, gf = 0.f, gg = 0.f, go = 0.f;                           \
        _Pragma("unroll")                                                       \
        for (int s = 0; s < 4; ++s) {                                           \
            gi += gbuf[s][0][bb][jj];                                           \
            gf += gbuf[s][1][bb][jj];                                           \
            gg += gbuf[s][2][bb][jj];                                           \
            go += gbuf[s][3][bb][jj];                                           \
        }                                                                       \
        float iv = sigf(gi + bsum[jg]);                                         \
        float fv = sigf(gf + bsum[512 + jg]);                                   \
        float gv = tanhf(gg + bsum[1024 + jg]);                                 \
        float ov = sigf(go + bsum[1536 + jg]);                                  \
        float c = fv * cbuf[bb * 512 + jg] + iv * gv;                           \
        cbuf[bb * 512 + jg] = c;                                                \
        float h = ov * tanhf(c);                                                \
        (H_STORE);                                                              \
    }

__global__ __launch_bounds__(512) void
enc_step(const unsigned short* __restrict__ Qbf,      // [64,512,256]
         const unsigned short* __restrict__ hprev, long long hstride,
         const unsigned short* __restrict__ W,        // [2048,768]
         const float* __restrict__ bsum,              // [2048]
         float* __restrict__ cbuf,                    // [64,512]
         unsigned short* __restrict__ enc_ext,        // [64,513,512]
         int t) {
    STEP_BODY(768, Qbf + (size_t)b * (512 * 256) + (size_t)t * 256, 256)
    CELL_UPDATE(enc_ext[((size_t)bb * 513 + t) * 512 + jg] = f2bf(h))
}

__global__ __launch_bounds__(512) void
dec_step(const unsigned short* __restrict__ startbf,  // [512]
         const unsigned short* __restrict__ enc_ext,  // [64,513,512]
         const int* __restrict__ tgt,                 // [64,512]
         const unsigned short* __restrict__ hprev, long long hstride,
         const unsigned short* __restrict__ W,        // [2048,1024]
         const float* __restrict__ bsum,
         float* __restrict__ cbuf,
         unsigned short* __restrict__ hnew,           // [64,512]
         unsigned char* __restrict__ mask,            // [64,513]
         int t) {
    // lazy mask update for step t-1 (scores of step t-1 already consumed it)
    if (blockIdx.x == 0 && threadIdx.x < 64 && t > 0) {
        int tg = tgt[threadIdx.x * 512 + (t - 1)];
        mask[threadIdx.x * 513 + tg] = 1;
    }
    STEP_BODY(1024,
              (t == 0) ? startbf
                       : enc_ext + ((size_t)b * 513 + tgt[b * 512 + (t - 1)]) * 512,
              512)
    CELL_UPDATE(hnew[bb * 512 + jg] = f2bf(h))
}

// --------------------------- e1 (hoisted) GEMM -----------------------------
// e1[32832,512] (bf16) = enc_ext[32832,512] @ W1^T + b1.
// Wave computes a 16x64 strip (A fragment reused across 4 B tiles).
__global__ __launch_bounds__(128) void
e1_gemm(const unsigned short* __restrict__ A,
        const unsigned short* __restrict__ W1,
        const float* __restrict__ b1,
        unsigned short* __restrict__ e1) {
    int wave = threadIdx.x >> 5, lane = threadIdx.x & 31;
    int job = blockIdx.x * 4 + wave;           // 0..16415
    int rowTile = job >> 3, cq = job & 7;
    int m0 = rowTile * 16, n0 = cq * 64;
    int ln = lane & 15, hi = lane >> 4;
    const unsigned short* arow = A + (size_t)(m0 + ln) * 512;
    const unsigned short* wb = W1 + (size_t)(n0 + ln) * 512;
    v8f c0 = {}, c1 = {}, c2 = {}, c3 = {};
#pragma unroll 1
    for (int kk = 0; kk < 512; kk += 32) {
        int kb = kk + hi * 8;
        FragBF a, b0, b1f, b2, b3;
        a.q[0]  = *(const uint4*)(arow + kb);
        a.q[1]  = *(const uint4*)(arow + kb + 16);
        b0.q[0] = *(const uint4*)(wb + kb);
        b0.q[1] = *(const uint4*)(wb + kb + 16);
        b1f.q[0] = *(const uint4*)(wb + 16 * 512 + kb);
        b1f.q[1] = *(const uint4*)(wb + 16 * 512 + kb + 16);
        b2.q[0] = *(const uint4*)(wb + 32 * 512 + kb);
        b2.q[1] = *(const uint4*)(wb + 32 * 512 + kb + 16);
        b3.q[0] = *(const uint4*)(wb + 48 * 512 + kb);
        b3.q[1] = *(const uint4*)(wb + 48 * 512 + kb + 16);
        c0 = wmma_bf(a, b0, c0);
        c1 = wmma_bf(a, b1f, c1);
        c2 = wmma_bf(a, b2, c2);
        c3 = wmma_bf(a, b3, c3);
    }
    v8f* accs[4] = {&c0, &c1, &c2, &c3};
#pragma unroll
    for (int i = 0; i < 4; ++i) {
        float bias = b1[n0 + i * 16 + ln];
        v8f& c = *accs[i];
#pragma unroll
        for (int p = 0; p < 8; ++p) {
            int M = hi ? p + 8 : p;
            e1[(size_t)(m0 + M) * 512 + n0 + i * 16 + ln] = f2bf(c[p] + bias);
        }
    }
}

// --------------------------- e2 GEMM (per dec step) ------------------------
__global__ __launch_bounds__(128) void
e2_gemm(const unsigned short* __restrict__ h,
        const unsigned short* __restrict__ W2,
        const float* __restrict__ b2,
        float* __restrict__ e2) {
    int wave = threadIdx.x >> 5, lane = threadIdx.x & 31;
    int job = blockIdx.x * 4 + wave;           // 0..31
    int rowTile = job >> 3, cq = job & 7;
    int m0 = rowTile * 16, n0 = cq * 64;
    int ln = lane & 15, hi = lane >> 4;
    const unsigned short* arow = h + (size_t)(m0 + ln) * 512;
    const unsigned short* wb = W2 + (size_t)(n0 + ln) * 512;
    v8f c0 = {}, c1 = {}, c2 = {}, c3 = {};
#pragma unroll 1
    for (int kk = 0; kk < 512; kk += 32) {
        int kb = kk + hi * 8;
        FragBF a, b0, b1f, b2, b3;
        a.q[0]  = *(const uint4*)(arow + kb);
        a.q[1]  = *(const uint4*)(arow + kb + 16);
        b0.q[0] = *(const uint4*)(wb + kb);
        b0.q[1] = *(const uint4*)(wb + kb + 16);
        b1f.q[0] = *(const uint4*)(wb + 16 * 512 + kb);
        b1f.q[1] = *(const uint4*)(wb + 16 * 512 + kb + 16);
        b2.q[0] = *(const uint4*)(wb + 32 * 512 + kb);
        b2.q[1] = *(const uint4*)(wb + 32 * 512 + kb + 16);
        b3.q[0] = *(const uint4*)(wb + 48 * 512 + kb);
        b3.q[1] = *(const uint4*)(wb + 48 * 512 + kb + 16);
        c0 = wmma_bf(a, b0, c0);
        c1 = wmma_bf(a, b1f, c1);
        c2 = wmma_bf(a, b2, c2);
        c3 = wmma_bf(a, b3, c3);
    }
    v8f* accs[4] = {&c0, &c1, &c2, &c3};
#pragma unroll
    for (int i = 0; i < 4; ++i) {
        float bias = b2[n0 + i * 16 + ln];
        v8f& c = *accs[i];
#pragma unroll
        for (int p = 0; p < 8; ++p) {
            int M = hi ? p + 8 : p;
            e2[(size_t)(m0 + M) * 512 + n0 + i * 16 + ln] = c[p] + bias;
        }
    }
}

// ---------------------- pointer-attention score reduce ---------------------
// scores[b,s] = sum_k tanh(e1[b,s,k] + e2[b,k]) * v[k]; masked fill; -> d_out
__global__ __launch_bounds__(256) void
attn_step(const unsigned short* __restrict__ e1,
          const float* __restrict__ e2,
          const float* __restrict__ vw,
          const unsigned char* __restrict__ mask,
          float* __restrict__ out, int t) {
    int wave = threadIdx.x >> 5, lane = threadIdx.x & 31;
    int row = blockIdx.x * 8 + wave;       // 0..32831  == b*513 + s
    int b = row / 513, s = row - b * 513;
    const unsigned short* er = e1 + (size_t)row * 512;
    const float* e2r = e2 + (size_t)b * 512;
    float acc = 0.f;
#pragma unroll
    for (int i = 0; i < 2; ++i) {
        int k = i * 256 + lane * 8;
        uint4 pk = *(const uint4*)(er + k);
        float4 ea = *(const float4*)(e2r + k);
        float4 eb = *(const float4*)(e2r + k + 4);
        float4 va = *(const float4*)(vw + k);
        float4 vb = *(const float4*)(vw + k + 4);
        acc += tanhf(bf2f((unsigned short)(pk.x & 0xFFFF)) + ea.x) * va.x;
        acc += tanhf(bf2f((unsigned short)(pk.x >> 16))    + ea.y) * va.y;
        acc += tanhf(bf2f((unsigned short)(pk.y & 0xFFFF)) + ea.z) * va.z;
        acc += tanhf(bf2f((unsigned short)(pk.y >> 16))    + ea.w) * va.w;
        acc += tanhf(bf2f((unsigned short)(pk.z & 0xFFFF)) + eb.x) * vb.x;
        acc += tanhf(bf2f((unsigned short)(pk.z >> 16))    + eb.y) * vb.y;
        acc += tanhf(bf2f((unsigned short)(pk.w & 0xFFFF)) + eb.z) * vb.z;
        acc += tanhf(bf2f((unsigned short)(pk.w >> 16))    + eb.w) * vb.w;
    }
#pragma unroll
    for (int off = 16; off; off >>= 1) acc += __shfl_xor(acc, off, 32);
    if (lane == 0) {
        float sc = mask[b * 513 + s] ? NEGV : acc;
        out[((size_t)b * 512 + t) * 513 + s] = sc;
    }
}

// ------------------------------- host driver -------------------------------

extern "C" void kernel_launch(void* const* d_in, const int* in_sizes, int n_in,
                              void* d_out, int out_size, void* d_ws, size_t ws_size,
                              hipStream_t stream) {
    (void)in_sizes; (void)n_in; (void)out_size; (void)ws_size;
    const float* Q        = (const float*)d_in[0];
    const float* enc_W_ih = (const float*)d_in[1];
    const float* enc_W_hh = (const float*)d_in[2];
    const float* enc_b_ih = (const float*)d_in[3];
    const float* enc_b_hh = (const float*)d_in[4];
    const float* dec_W_ih = (const float*)d_in[5];
    const float* dec_W_hh = (const float*)d_in[6];
    const float* dec_b_ih = (const float*)d_in[7];
    const float* dec_b_hh = (const float*)d_in[8];
    const float* W1_w     = (const float*)d_in[9];
    const float* W1_b     = (const float*)d_in[10];
    const float* W2_w     = (const float*)d_in[11];
    const float* W2_b     = (const float*)d_in[12];
    const float* v_w      = (const float*)d_in[13];
    const float* dstart   = (const float*)d_in[14];
    const int*   tgt      = (const int*)d_in[15];
    float* out = (float*)d_out;

    // ---- workspace carve (all 256B aligned) ----
    char* p = (char*)d_ws;
    auto alloc = [&](size_t bytes) -> char* {
        char* r = p;
        p += (bytes + 255) & ~(size_t)255;
        return r;
    };
    unsigned short* Wenc   = (unsigned short*)alloc((size_t)2048 * 768 * 2);
    unsigned short* Wdec   = (unsigned short*)alloc((size_t)2048 * 1024 * 2);
    unsigned short* W1bf   = (unsigned short*)alloc((size_t)512 * 512 * 2);
    unsigned short* W2bf   = (unsigned short*)alloc((size_t)512 * 512 * 2);
    unsigned short* Qbf    = (unsigned short*)alloc((size_t)64 * 512 * 256 * 2);
    unsigned short* encext = (unsigned short*)alloc((size_t)64 * 513 * 512 * 2);
    unsigned short* e1bf   = (unsigned short*)alloc((size_t)64 * 513 * 512 * 2);
    float*          bsumE  = (float*)alloc(2048 * 4);
    float*          bsumD  = (float*)alloc(2048 * 4);
    float*          cbuf   = (float*)alloc((size_t)64 * 512 * 4);
    unsigned short* hbuf0  = (unsigned short*)alloc((size_t)64 * 512 * 2);
    unsigned short* hbuf1  = (unsigned short*)alloc((size_t)64 * 512 * 2);
    unsigned short* hzero  = (unsigned short*)alloc((size_t)64 * 512 * 2);
    float*          e2buf  = (float*)alloc((size_t)64 * 512 * 4);
    unsigned short* startb = (unsigned short*)alloc(512 * 2);
    unsigned char*  mask   = (unsigned char*)alloc(64 * 513 + 64);

    // ---- one-time conversions / init (re-run every call: deterministic) ----
    cvt_bf<<<8192, 256, 0, stream>>>(Qbf, Q, (size_t)64 * 512 * 256);
    cvt_bf<<<512, 256, 0, stream>>>(W1bf, W1_w, (size_t)512 * 512);
    cvt_bf<<<512, 256, 0, stream>>>(W2bf, W2_w, (size_t)512 * 512);
    cvt_bf<<<2, 256, 0, stream>>>(startb, dstart, 512);
    cat_cvt<<<2048, 256, 0, stream>>>(Wenc, enc_W_ih, enc_W_hh, 2048, 256, 512);
    cat_cvt<<<2048, 256, 0, stream>>>(Wdec, dec_W_ih, dec_W_hh, 2048, 512, 512);
    vadd_f32<<<8, 256, 0, stream>>>(bsumE, enc_b_ih, enc_b_hh, 2048);
    vadd_f32<<<8, 256, 0, stream>>>(bsumD, dec_b_ih, dec_b_hh, 2048);
    zero_u32<<<128, 256, 0, stream>>>((unsigned int*)cbuf, (size_t)64 * 512);
    zero_u32<<<64, 256, 0, stream>>>((unsigned int*)hzero, (size_t)64 * 512 / 2);
    zero_u32<<<64, 256, 0, stream>>>((unsigned int*)mask, (size_t)(64 * 513 + 64) / 4);
    zero_pad<<<128, 256, 0, stream>>>(encext);

    // ---- encoder recurrence: 512 fused LSTM-step launches ----
    for (int t = 0; t < 512; ++t) {
        const unsigned short* hprev = (t == 0) ? hzero : (encext + (size_t)(t - 1) * 512);
        long long hstride = (t == 0) ? 512LL : 513LL * 512LL;
        enc_step<<<32, 512, 0, stream>>>(Qbf, hprev, hstride, Wenc, bsumE, cbuf, encext, t);
    }

    // ---- hoisted e1 = enc_ext @ W1^T + b1 (big parallel GEMM, bf16 out) ----
    e1_gemm<<<4104, 128, 0, stream>>>(encext, W1bf, W1_b, e1bf);

    // ---- decoder: LSTM -> e2 -> masked pointer scores, per step ----
    for (int t = 0; t < 512; ++t) {
        const unsigned short* hprev;
        long long hstride;
        if (t == 0) { hprev = encext + (size_t)511 * 512; hstride = 513LL * 512LL; }
        else        { hprev = (t & 1) ? hbuf0 : hbuf1;    hstride = 512LL; }
        unsigned short* hnew = (t & 1) ? hbuf1 : hbuf0;
        dec_step<<<32, 512, 0, stream>>>(startb, encext, tgt, hprev, hstride,
                                         Wdec, bsumD, cbuf, hnew, mask, t);
        e2_gemm<<<8, 128, 0, stream>>>(hnew, W2bf, W2_b, e2buf);
        attn_step<<<4104, 256, 0, stream>>>(e1bf, e2buf, v_w, mask, out, t);
    }
}
